// UnsupervisedMPNN_309237645658
// MI455X (gfx1250) — matrix operand order repaired and checked by reading, not compiled, not run on hardware
//
#include <hip/hip_runtime.h>
#include <stdint.h>

#define NODES   100000
#define EDGES   400000
#define DDIM    32
#define EHID    128
#define MPSTEPS 6

typedef __attribute__((ext_vector_type(2)))  float  v2f;
typedef __attribute__((ext_vector_type(8)))  float  v8f;
typedef __attribute__((ext_vector_type(16))) __bf16 v16bf;
typedef __attribute__((ext_vector_type(4)))  unsigned int v4u;
typedef __attribute__((ext_vector_type(8)))  int    v8i;
typedef __attribute__((ext_vector_type(4)))  int    v4i;

#if __has_builtin(__builtin_amdgcn_wmma_f32_16x16x4_f32)
#define HAVE_WMMA_F32 1
#else
#define HAVE_WMMA_F32 0
#endif

#if __has_builtin(__builtin_amdgcn_tensor_load_to_lds) && __has_builtin(__builtin_amdgcn_s_wait_tensorcnt)
#define HAVE_TDM 1
#else
#define HAVE_TDM 0
#endif

static __device__ __forceinline__ uint16_t f2bf(float f) {
  union { float f; uint32_t i; } c; c.f = f;
  uint32_t r = 0x7FFFu + ((c.i >> 16) & 1u);           // round-to-nearest-even
  return (uint16_t)((c.i + r) >> 16);
}
static __device__ __forceinline__ __bf16 f2bfh(float f) {
  uint16_t b = f2bf(f);
  return __builtin_bit_cast(__bf16, b);
}

// ---- C = A(16x32) * B(32x16) + C, bf16 inputs, f32 accum (one WMMA) ----
// A 16-bit 16x32 layout: lane(m,g), slot s -> k = (v&4?16:0) + 8g + 2(v&3) + w
// B 16-bit 32x16 layout: lane(n,g), slot s -> k = (v&4?16:0) + 8g + 2(v&3) + w
template <typename FA, typename FB>
static __device__ __forceinline__ v8f wmma_bf16_k32(int lane, FA loadA, FB loadB, v8f c) {
  const int mn = lane & 15, g = lane >> 4;
  v16bf a, b;
#pragma unroll
  for (int s = 0; s < 16; ++s) {
    const int v = s >> 1, w = s & 1;
    const int ka = ((v & 4) ? 16 : 0) + 8 * g + 2 * (v & 3) + w;
    a[s] = f2bfh(loadA(mn, ka));
    const int kb = ((v & 4) ? 16 : 0) + 8 * g + 2 * (v & 3) + w;
    b[s] = f2bfh(loadB(kb, mn));
  }
  return __builtin_amdgcn_wmma_f32_16x16x32_bf16(false, a, false, b, (short)0, c, false, false);
}

// ---- C = A(16x32) * B(32x16) + C in pure f32 via 8x V_WMMA_F32_16X16X4_F32 ----
template <typename FA, typename FB>
static __device__ __forceinline__ v8f wmma_f32_k32(int lane, FA loadA, FB loadB, v8f c) {
#if HAVE_WMMA_F32
  const int mn = lane & 15, g = lane >> 4;
#pragma unroll
  for (int kc = 0; kc < 8; ++kc) {
    v2f a, b;
#pragma unroll
    for (int v = 0; v < 2; ++v) {
      const int k = kc * 4 + v + 2 * g;
      a[v] = loadA(mn, k);
      b[v] = loadB(k, mn);
    }
    c = __builtin_amdgcn_wmma_f32_16x16x4_f32(false, a, false, b, (short)0, c, false, false);
  }
  return c;
#else
  return wmma_bf16_k32(lane, loadA, loadB, c);
#endif
}

// ---------------- lin0: X = relu(n_feat @ lin0_w.T + b), f32 WMMA ----------------
__global__ void k_lin0(const float* __restrict__ nf, const float* __restrict__ w,
                       const float* __restrict__ bias, float* __restrict__ X) {
  const int wave = (int)((blockIdx.x * blockDim.x + threadIdx.x) >> 5);
  const int lane = threadIdx.x & 31;
  if (wave >= NODES / 16) return;
  const int base = wave * 16;
  auto la = [&](int m, int k) { return nf[(base + m) * DDIM + k]; };
  v8f c0 = {}, c1 = {};
  c0 = wmma_f32_k32(lane, la, [&](int k, int n) { return w[n * DDIM + k]; }, c0);
  c1 = wmma_f32_k32(lane, la, [&](int k, int n) { return w[(16 + n) * DDIM + k]; }, c1);
  const int g = lane >> 4, n0 = lane & 15;
#pragma unroll
  for (int v = 0; v < 8; ++v) {
    const int row = base + v + 8 * g;
    float y0 = c0[v] + bias[n0];
    float y1 = c1[v] + bias[16 + n0];
    X[row * DDIM + n0]      = y0 > 0.f ? y0 : 0.f;
    X[row * DDIM + 16 + n0] = y1 > 0.f ? y1 : 0.f;
  }
}

// ---- en1: eh = relu(e_feat @ en1_w.T + b), bf16 WMMA; output written in the
//      A-fragment swizzle k_en2 consumes (contiguous 32B per lane per k-chunk) ----
__global__ void k_en1(const float* __restrict__ ef, const float* __restrict__ w1,
                      const float* __restrict__ b1, uint16_t* __restrict__ ehf) {
  const int sub = (int)((blockIdx.x * blockDim.x + threadIdx.x) >> 5);
  const int lane = threadIdx.x & 31;
  if (sub >= EDGES / 16) return;
  const int base = sub * 16;
  const int g = lane >> 4, n0 = lane & 15;
  auto la = [&](int m, int k) { return ef[(base + m) * DDIM + k]; };
#pragma unroll
  for (int nt = 0; nt < 8; ++nt) {
    v8f c = {};
    c = wmma_bf16_k32(lane, la,
                      [&](int k, int n) { return w1[(nt * 16 + n) * DDIM + k]; }, c);
#pragma unroll
    for (int v = 0; v < 8; ++v) {
      const int mp = v + 8 * g;        // edge row within tile
      const int kp = nt * 16 + n0;     // feature -> K of next GEMM
      float y = c[v] + b1[kp];
      y = y > 0.f ? y : 0.f;
      // invert the A 16-bit 16x32 layout for (mp, kp)
      const int kc = kp >> 5, kk = kp & 31;
      const int vhi = (kk >> 4) & 1, kr = kk & 15;
      const int gg = kr >> 3, vlo = (kr & 7) >> 1, ww = kr & 1;
      const int slot = ((((vhi << 2) | vlo) << 1) | ww);
      ehf[(size_t)((((sub * 4 + kc) * 32) + (gg * 16 + mp)) * 16 + slot)] = f2bf(y);
    }
  }
}

// ---- prep: pre-swizzle en2_w (1024x128 f32) into bf16 B-fragment order so the
//      LDS panel fill is a flat 256KB copy the TDM can DMA ----
__global__ void k_prep(const float* __restrict__ w2, uint16_t* __restrict__ panel) {
  const int idx = blockIdx.x * blockDim.x + threadIdx.x;
  if (idx >= 64 * 4 * 32 * 16) return;
  const int s  = idx & 15;
  const int ln = (idx >> 4) & 31;
  const int kc = (idx >> 9) & 3;
  const int nt = idx >> 11;
  const int k = kc * 32 + ((s >> 3) << 4) + 8 * (ln >> 4) + (s & 7);   // B 16-bit 32x16 layout
  const int n = nt * 16 + (ln & 15);
  panel[idx] = f2bf(w2[n * EHID + k]);          // B[k][n] = en2_w[n][k]
}

// ---- en2: w_e(bf16) = eh @ en2_w.T + b.  256KB bf16 B-panel DMA'd into LDS by
//      the Tensor Data Mover (320KB/WGP makes this possible); fragments read as
//      wide ds loads; n-tiles processed 4-at-a-time so the 0.82GB w_e stream is
//      written as full u32 (lo/hi bf16 pairs), 128B contiguous per wave-row. ----
__global__ void k_en2(const uint16_t* __restrict__ ehf, const uint16_t* __restrict__ panel,
                      const float* __restrict__ b2, uint16_t* __restrict__ wbf) {
  extern __shared__ __attribute__((aligned(32))) uint16_t ldsB[];  // [nt][kc][lane][slot] 64*4*32*16
#if HAVE_TDM
  if (threadIdx.x < 32) {           // one wave issues the TDM descriptor
    const uint64_t ga = (uint64_t)(uintptr_t)panel;
    v4u g0 = {};
    g0[0] = 1u;                                              // count=1 valid descriptor
    g0[1] = 0u;                                              // lds_addr = 0 (start of our LDS)
    g0[2] = (uint32_t)ga;                                    // global_addr[31:0]
    g0[3] = (uint32_t)((ga >> 32) & 0x01FFFFFFu) | (2u << 30); // global_addr[56:32] | type=2
    v8i g1 = {};
    g1[0] = (1 << 16);                                       // data_size=1 (2B); no mask/iterate/pad
    g1[1] = (int)(8192u << 16);                              // tensor_dim0 = 8192 (lo16)
    g1[2] = (16 << 16);                                      // tensor_dim0 hi=0 | tensor_dim1=16
    g1[3] = (int)(8192u << 16);                              // tensor_dim1 hi=0 | tile_dim0=8192
    g1[4] = 16;                                              // tile_dim1=16, tile_dim2=0
    g1[5] = 8192;                                            // tensor_dim0_stride = 8192
    g1[6] = 0;                                               // stride0 hi | stride1 lo16
    g1[7] = 2;                                               // tensor_dim1_stride = 131072 (bits 47:16)
    v4i g2 = {}, g3 = {};
    v8i g4 = {};                                             // unused trailing group (clang-23 6-arg form)
    __builtin_amdgcn_tensor_load_to_lds(g0, g1, g2, g3, g4, 0);
    __builtin_amdgcn_s_wait_tensorcnt((short)0);
  }
  __syncthreads();
#else
  for (int idx = threadIdx.x; idx < 64 * 4 * 32 * 16; idx += blockDim.x)
    ldsB[idx] = panel[idx];
  __syncthreads();
#endif
  const int lane = threadIdx.x & 31;
  const int wv = threadIdx.x >> 5;
  const int g = lane >> 4, n0 = lane & 15;
  const int wpb = blockDim.x >> 5;
  uint32_t* __restrict__ wp32 = (uint32_t*)wbf;
  for (int sub = blockIdx.x * wpb + wv; sub < EDGES / 16; sub += (int)gridDim.x * wpb) {
    v16bf a[4];
#pragma unroll
    for (int kc = 0; kc < 4; ++kc)
      a[kc] = *(const v16bf*)&ehf[(size_t)(((sub * 4 + kc) * 32 + lane) * 16)];
    for (int p = 0; p < 16; ++p) {                 // 4 n-tiles per iteration
      v8f c[4];
#pragma unroll
      for (int q = 0; q < 4; ++q) {
        const int nt = 4 * p + q;
        v8f acc = {};
#pragma unroll
        for (int kc = 0; kc < 4; ++kc) {
          v16bf b = *(const v16bf*)&ldsB[((nt * 4 + kc) * 32 + lane) * 16];
          acc = __builtin_amdgcn_wmma_f32_16x16x32_bf16(false, a[kc], false, b, (short)0, acc, false, false);
        }
        c[q] = acc;
      }
      // tiles (4p,4p+2) form lo/hi of pair i2=p at o=n0; (4p+1,4p+3) at o=16+n0
#pragma unroll
      for (int v = 0; v < 8; ++v) {
        const int e2 = sub * 16 + v + 8 * g;
        const size_t rowbase = ((size_t)e2 * 16 + p) * 32;
        const uint32_t lo0 = f2bf(c[0][v] + b2[(4 * p + 0) * 16 + n0]);
        const uint32_t hi0 = f2bf(c[2][v] + b2[(4 * p + 2) * 16 + n0]);
        wp32[rowbase + n0] = lo0 | (hi0 << 16);
        const uint32_t lo1 = f2bf(c[1][v] + b2[(4 * p + 1) * 16 + n0]);
        const uint32_t hi1 = f2bf(c[3][v] + b2[(4 * p + 3) * 16 + n0]);
        wp32[rowbase + 16 + n0] = lo1 | (hi1 << 16);
      }
    }
  }
}

// ---- msg: one wave per edge; gather x[src], per-edge 32x32 bf16 matvec in f32,
//      atomic scatter-add into agg (segment_sum) ----
__global__ void k_msg(const float* __restrict__ X, const uint16_t* __restrict__ wbf,
                      const int* __restrict__ src, const int* __restrict__ dst,
                      float* __restrict__ agg) {
  __shared__ float xs[8][DDIM];
  const int lane = threadIdx.x & 31;
  const int wv = threadIdx.x >> 5;
  const int wpb = blockDim.x >> 5;
  const int stride = (int)gridDim.x * wpb;
  for (int e = blockIdx.x * wpb + wv; e < EDGES; e += stride) {
    const int s = src[e];
    const int d = dst[e];
    xs[wv][lane] = X[s * DDIM + lane];
    float acc = 0.f;
    const uint32_t* wp = (const uint32_t*)wbf + (size_t)e * 512 + lane;
#pragma unroll
    for (int i2 = 0; i2 < 16; ++i2) {
      const uint32_t u = wp[(size_t)i2 * 32];            // lanes contiguous: 128B/wave
      const float wlo = __builtin_bit_cast(float, u << 16);
      const float whi = __builtin_bit_cast(float, u & 0xFFFF0000u);
      acc = fmaf(xs[wv][2 * i2 + 0], wlo, acc);
      acc = fmaf(xs[wv][2 * i2 + 1], whi, acc);
    }
    atomicAdd(&agg[d * DDIM + lane], acc);
  }
}

// ---- GRU: m = relu(agg + conv_bias); gi = m@wih.T, gh = h@whh.T via f32 WMMA;
//      gates; in-place h update (each wave owns its 16 nodes) ----
__global__ void k_gru(float* __restrict__ X, const float* __restrict__ agg,
                      const float* __restrict__ cb, const float* __restrict__ wih,
                      const float* __restrict__ whh, const float* __restrict__ bih,
                      const float* __restrict__ bhh) {
  const int wave = (int)((blockIdx.x * blockDim.x + threadIdx.x) >> 5);
  const int lane = threadIdx.x & 31;
  if (wave >= NODES / 16) return;
  const int base = wave * 16;
  auto lam = [&](int m, int k) {
    float t = agg[(base + m) * DDIM + k] + cb[k];
    return t > 0.f ? t : 0.f;
  };
  auto lah = [&](int m, int k) { return X[(base + m) * DDIM + k]; };
  v8f cI[6], cH[6];
#pragma unroll
  for (int nt = 0; nt < 6; ++nt) {
    v8f zi = {}, zh = {};
    cI[nt] = wmma_f32_k32(lane, lam, [&](int k, int n) { return wih[(nt * 16 + n) * DDIM + k]; }, zi);
    cH[nt] = wmma_f32_k32(lane, lah, [&](int k, int n) { return whh[(nt * 16 + n) * DDIM + k]; }, zh);
  }
  const int g = lane >> 4, n0 = lane & 15;
#pragma unroll
  for (int t = 0; t < 2; ++t) {
#pragma unroll
    for (int v = 0; v < 8; ++v) {
      const int row = base + v + 8 * g;
      const int col = t * 16 + n0;
      const float gr = cI[t][v] + bih[col] + cH[t][v] + bhh[col];
      const float gz = cI[2 + t][v] + bih[32 + col] + cH[2 + t][v] + bhh[32 + col];
      const float r = 1.f / (1.f + __expf(-gr));
      const float z = 1.f / (1.f + __expf(-gz));
      const float n = tanhf(cI[4 + t][v] + bih[64 + col] + r * (cH[4 + t][v] + bhh[64 + col]));
      const float hold = X[row * DDIM + col];
      X[row * DDIM + col] = (1.f - z) * n + z * hold;
    }
  }
}

extern "C" void kernel_launch(void* const* d_in, const int* in_sizes, int n_in,
                              void* d_out, int out_size, void* d_ws, size_t ws_size,
                              hipStream_t stream) {
  (void)in_sizes; (void)n_in; (void)ws_size;
  const float* nf  = (const float*)d_in[0];
  const float* ef  = (const float*)d_in[1];
  const int*   src = (const int*)d_in[2];
  const int*   dst = (const int*)d_in[3];
  const float* l0w = (const float*)d_in[4];
  const float* l0b = (const float*)d_in[5];
  const float* e1w = (const float*)d_in[6];
  const float* e1b = (const float*)d_in[7];
  const float* e2w = (const float*)d_in[8];
  const float* e2b = (const float*)d_in[9];
  const float* cb  = (const float*)d_in[10];
  const float* wih = (const float*)d_in[11];
  const float* whh = (const float*)d_in[12];
  const float* bih = (const float*)d_in[13];
  const float* bhh = (const float*)d_in[14];

  // ws: X 12.8MB | agg 12.8MB | panel 256KB (swizzled bf16 en2_w) | ehf 102.4MB | wbf 819.2MB
  char* ws = (char*)d_ws;
  float*    X     = (float*)ws;
  float*    agg   = (float*)(ws + ((size_t)16 << 20));
  uint16_t* panel = (uint16_t*)(ws + ((size_t)30 << 20));
  uint16_t* ehf   = (uint16_t*)(ws + ((size_t)32 << 20));
  uint16_t* wbf   = (uint16_t*)(ws + ((size_t)144 << 20));

  const int nodeBlocks = (NODES / 16 * 32 + 255) / 256;   // 782
  const int edgeBlocks = (EDGES / 16 * 32 + 255) / 256;   // 3125

  k_lin0<<<nodeBlocks, 256, 0, stream>>>(nf, l0w, l0b, X);
  k_en1<<<edgeBlocks, 256, 0, stream>>>(ef, e1w, e1b, ehf);
  k_prep<<<512, 256, 0, stream>>>(e2w, panel);
  k_en2<<<1024, 256, 256 * 1024, stream>>>(ehf, panel, e2b, wbf);

  for (int s = 0; s < MPSTEPS; ++s) {
    (void)hipMemsetAsync(agg, 0, (size_t)NODES * DDIM * sizeof(float), stream);
    k_msg<<<2048, 256, 0, stream>>>(X, wbf, src, dst, agg);
    k_gru<<<nodeBlocks, 256, 0, stream>>>(X, agg, cb, wih, whh, bih, bhh);
  }
  (void)hipMemcpyAsync(d_out, X, (size_t)out_size * sizeof(float), hipMemcpyDeviceToDevice, stream);
}
